// Pointnet2MSG_6691559047648
// MI455X (gfx1250) — matrix operand order repaired and checked by reading, not compiled, not run on hardware
//
#include <hip/hip_runtime.h>
#include <hip/hip_bf16.h>

// ---------------------------------------------------------------------------
// Types for CDNA5 WMMA (gfx1250, wave32)
// ---------------------------------------------------------------------------
typedef __attribute__((ext_vector_type(16))) __bf16       v16bf;
typedef __attribute__((ext_vector_type(8)))  float        v8f;
typedef __attribute__((ext_vector_type(4)))  unsigned int v4u;

union BFrag { v16bf v; v4u u[2]; };

// ---------------------------------------------------------------------------
// Elementwise / data-movement kernels
// ---------------------------------------------------------------------------
__global__ void split_pc_kernel(const float* __restrict__ pc, int tot,
                                float* __restrict__ xyz, __bf16* __restrict__ feats) {
  int i = blockIdx.x * blockDim.x + threadIdx.x;
  if (i >= tot) return;
  const float* s = pc + (size_t)i * 9;
  float* x = xyz + (size_t)i * 3;
  __bf16* f = feats + (size_t)i * 6;
  x[0] = s[0]; x[1] = s[1]; x[2] = s[2];
#pragma unroll
  for (int c = 0; c < 6; ++c) f[c] = (__bf16)s[3 + c];
}

// Farthest point sampling: one block per batch, sequential argmax selection.
#define FPS_T 256
__global__ void fps_kernel(const float* __restrict__ xyz, int Np, int npoint,
                           int* __restrict__ out_idx, float* __restrict__ dist) {
  int b = blockIdx.x;
  const float* px = xyz + (size_t)b * Np * 3;
  float* pd = dist + (size_t)b * Np;
  int t = threadIdx.x;
  for (int i = t; i < Np; i += FPS_T) pd[i] = 1e10f;
  __shared__ float s_val[FPS_T];
  __shared__ int   s_idx[FPS_T];
  __shared__ float sfx, sfy, sfz;
  __syncthreads();
  int far = 0;
  for (int it = 0; it < npoint; ++it) {
    if (t == 0) {
      out_idx[b * npoint + it] = far;             // emit BEFORE update (lax.scan order)
      sfx = px[far * 3 + 0]; sfy = px[far * 3 + 1]; sfz = px[far * 3 + 2];
    }
    __syncthreads();
    float fx = sfx, fy = sfy, fz = sfz;
    float bestv = -1.f; int besti = 0;
    for (int i = t; i < Np; i += FPS_T) {
      float dx = px[i * 3 + 0] - fx;
      float dy = px[i * 3 + 1] - fy;
      float dz = px[i * 3 + 2] - fz;
      float d = dx * dx + dy * dy + dz * dz;
      float nd = fminf(pd[i], d);
      pd[i] = nd;
      if (nd > bestv) { bestv = nd; besti = i; }
    }
    s_val[t] = bestv; s_idx[t] = besti;
    __syncthreads();
    for (int s = FPS_T >> 1; s > 0; s >>= 1) {
      if (t < s) {
        float ov = s_val[t + s]; int oi = s_idx[t + s];
        if (ov > s_val[t] || (ov == s_val[t] && oi < s_idx[t])) { s_val[t] = ov; s_idx[t] = oi; }
      }
      __syncthreads();
    }
    far = s_idx[0];
    __syncthreads();
  }
}

__global__ void gather_xyz_kernel(const float* __restrict__ xyz, const int* __restrict__ fidx,
                                  int Np, int S, int tot, float* __restrict__ nxyz) {
  int i = blockIdx.x * blockDim.x + threadIdx.x;
  if (i >= tot) return;
  int b = i / S;
  int src = fidx[i];
  const float* p = xyz + ((size_t)b * Np + src) * 3;
  nxyz[(size_t)i * 3 + 0] = p[0];
  nxyz[(size_t)i * 3 + 1] = p[1];
  nxyz[(size_t)i * 3 + 2] = p[2];
}

// Ball query: first nsample indices (ascending) with d2 < r2; pad with first hit.
__global__ void ball_query_kernel(const float* __restrict__ xyz, const float* __restrict__ nxyz,
                                  int Np, int S, int tot, float r2, int nsample,
                                  int* __restrict__ out) {
  int i = blockIdx.x * blockDim.x + threadIdx.x;
  if (i >= tot) return;
  int b = i / S;
  const float* px = xyz + (size_t)b * Np * 3;
  float qx = nxyz[(size_t)i * 3 + 0];
  float qy = nxyz[(size_t)i * 3 + 1];
  float qz = nxyz[(size_t)i * 3 + 2];
  int* o = out + (size_t)i * nsample;
  int cnt = 0, first = 0;
  for (int j = 0; j < Np && cnt < nsample; ++j) {
    float dx = px[j * 3 + 0] - qx;
    float dy = px[j * 3 + 1] - qy;
    float dz = px[j * 3 + 2] - qz;
    if (dx * dx + dy * dy + dz * dz < r2) {
      if (cnt == 0) first = j;
      o[cnt++] = j;
    }
  }
  for (; cnt < nsample; ++cnt) o[cnt] = first;
}

// Build SA GEMM input: rows=(b,s,j), cols = [rel_xyz(3) | feats(C) | zero pad].
__global__ void group_gather_kernel(const float* __restrict__ xyz, const float* __restrict__ nxyz,
                                    const __bf16* __restrict__ feats, int C,
                                    int Np, int S, int nsample, int KP,
                                    const int* __restrict__ idx, __bf16* __restrict__ A,
                                    long long tot) {
  long long g = (long long)blockIdx.x * blockDim.x + threadIdx.x;
  if (g >= tot) return;
  int col = (int)(g % KP);
  long long row = g / KP;
  int q = (int)(row / nsample);      // b*S + s
  int b = q / S;
  int pi = idx[row];
  float v = 0.f;
  if (col < 3) {
    v = xyz[((size_t)b * Np + pi) * 3 + col] - nxyz[(size_t)q * 3 + col];
  } else if (col < 3 + C) {
    v = (float)feats[((size_t)b * Np + pi) * C + (col - 3)];
  }
  A[(size_t)row * KP + col] = (__bf16)v;
}

// Max over nsample neighbors, write into concat feature buffer at channel offset.
__global__ void maxpool_kernel(const __bf16* __restrict__ A, int S, int nsample, int NPADw,
                               int cout, __bf16* __restrict__ outF, int Ctot, int coff, int tot) {
  int i = blockIdx.x * blockDim.x + threadIdx.x;
  if (i >= tot) return;
  int c = i % cout;
  int q = i / cout;                  // b*S + s
  size_t rb = (size_t)q * nsample;
  float m = -3.4e38f;
  for (int j = 0; j < nsample; ++j)
    m = fmaxf(m, (float)A[(rb + j) * NPADw + c]);
  outF[(size_t)q * Ctot + coff + c] = (__bf16)m;
}

// 3-NN with inverse-squared-distance weights (matches 1/(d2+1e-8), normalized).
__global__ void three_nn_kernel(const float* __restrict__ unk, const float* __restrict__ kn,
                                int Nu, int Nk, int tot,
                                int* __restrict__ idx3, float* __restrict__ w3) {
  int i = blockIdx.x * blockDim.x + threadIdx.x;
  if (i >= tot) return;
  int b = i / Nu;
  float qx = unk[(size_t)i * 3 + 0];
  float qy = unk[(size_t)i * 3 + 1];
  float qz = unk[(size_t)i * 3 + 2];
  const float* pk = kn + (size_t)b * Nk * 3;
  float d0 = 1e30f, d1 = 1e30f, d2 = 1e30f;
  int i0 = 0, i1 = 0, i2 = 0;
  for (int j = 0; j < Nk; ++j) {
    float dx = pk[j * 3 + 0] - qx;
    float dy = pk[j * 3 + 1] - qy;
    float dz = pk[j * 3 + 2] - qz;
    float d = dx * dx + dy * dy + dz * dz;
    if (d < d0)      { d2 = d1; i2 = i1; d1 = d0; i1 = i0; d0 = d; i0 = j; }
    else if (d < d1) { d2 = d1; i2 = i1; d1 = d;  i1 = j; }
    else if (d < d2) { d2 = d;  i2 = j; }
  }
  float w0 = 1.f / (d0 + 1e-8f), w1 = 1.f / (d1 + 1e-8f), w2 = 1.f / (d2 + 1e-8f);
  float s = w0 + w1 + w2;
  idx3[(size_t)i * 3 + 0] = i0; idx3[(size_t)i * 3 + 1] = i1; idx3[(size_t)i * 3 + 2] = i2;
  w3[(size_t)i * 3 + 0] = w0 / s; w3[(size_t)i * 3 + 1] = w1 / s; w3[(size_t)i * 3 + 2] = w2 / s;
}

// Build FP GEMM input: [interp(Ck) | unknown_feats(Cu) | zero pad].
__global__ void fp_input_kernel(const int* __restrict__ idx3, const float* __restrict__ w3,
                                const __bf16* __restrict__ knF, int Ck,
                                const __bf16* __restrict__ unkF, int Cu,
                                int Nu, int Nk, int KP, __bf16* __restrict__ A, long long tot) {
  long long g = (long long)blockIdx.x * blockDim.x + threadIdx.x;
  if (g >= tot) return;
  int col = (int)(g % KP);
  long long row = g / KP;            // b*Nu + n
  int b = (int)(row / Nu);
  float v = 0.f;
  if (col < Ck) {
    size_t kb = (size_t)b * Nk;
#pragma unroll
    for (int j = 0; j < 3; ++j) {
      int id = idx3[(size_t)row * 3 + j];
      float w = w3[(size_t)row * 3 + j];
      v += w * (float)knF[(kb + id) * Ck + col];
    }
  } else if (col < Ck + Cu) {
    v = (float)unkF[(size_t)row * Cu + (col - Ck)];
  }
  A[(size_t)row * KP + col] = (__bf16)v;
}

// Convert + pad weight (cout x cin f32) -> (ru32(cout) x ru32(cin) bf16).
__global__ void pad_weight_kernel(const float* __restrict__ W, int cin, int cout,
                                  int KP, int NPd, __bf16* __restrict__ Wp) {
  int i = blockIdx.x * blockDim.x + threadIdx.x;
  int tot = KP * NPd;
  if (i >= tot) return;
  int k = i % KP, n = i / KP;
  float v = (n < cout && k < cin) ? W[(size_t)n * cin + k] : 0.f;
  Wp[(size_t)n * KP + k] = (__bf16)v;
}

// Final (B,N,C) bf16 -> (B,C,N) f32.
__global__ void transpose_out_kernel(const __bf16* __restrict__ f, int N, int C,
                                     float* __restrict__ out, int tot) {
  int i = blockIdx.x * blockDim.x + threadIdx.x;
  if (i >= tot) return;
  int n = i % N;
  int c = (i / N) % C;
  int b = i / (N * C);
  out[i] = (float)f[((size_t)b * N + n) * C + c];
}

// ---------------------------------------------------------------------------
// Fused WMMA GEMM: Out[m,n] = relu(sum_k A[m,k]*W[n,k] * sc[n] + sh[n])
// A: M x KPAD bf16 row-major. Wp: NPAD x KPAD bf16 row-major (padded).
// Each wave computes a 32x32 output tile: 2 A-frags x 2 B-frags -> 4
// v_wmma_f32_16x16x32_bf16 per K-step (2x the reuse of a 16x32 tile).
// Fragment layouts per CDNA5 ISA 7.12.2 (16-bit A 16x32; B rows striped).
// Working sets fit in the 192MB L2, so fragments load straight from global
// (L2-resident) with WGP-scope prefetch hints instead of LDS staging.
// ---------------------------------------------------------------------------
__global__ __launch_bounds__(128)
void wmma_mlp_kernel(const __bf16* __restrict__ A, const __bf16* __restrict__ Wp,
                     const float* __restrict__ sc, const float* __restrict__ sh,
                     __bf16* __restrict__ Out,
                     int M, int Nout, int NPAD, int KPAD) {
  const int lane = threadIdx.x & 31;
  const int wave = blockIdx.x * (blockDim.x >> 5) + (threadIdx.x >> 5);
  const int tilesN = NPAD >> 5;
  const int tilesM = M >> 5;                    // M is always a multiple of 32 here
  if (wave >= tilesM * tilesN) return;
  const int tm = wave / tilesN;
  const int tn = wave - tm * tilesN;
  const int m0 = tm << 5;
  const int n0 = tn << 5;

  // A-frag: lanes 0-15 row m0+l, K{0..7,16..23}; lanes 16-31 K{8..15,24..31}.
  const __bf16* Ap0 = A + (size_t)(m0 + (lane & 15)) * KPAD + ((lane >> 4) << 3);
  const __bf16* Ap1 = Ap0 + (size_t)16 * KPAD;
  // B-frag (B[k][n] = W[n][k]): lane handles column n, K{0..7,16..23} / K{8..15,24..31}.
  const __bf16* Bp0 = Wp + (size_t)(n0 + (lane & 15)) * KPAD + ((lane >> 4) << 3);
  const __bf16* Bp1 = Bp0 + (size_t)16 * KPAD;

  union Acc { v8f v; float f[8]; };
  Acc a00, a01, a10, a11;
  a00.v = (v8f){0.f, 0.f, 0.f, 0.f, 0.f, 0.f, 0.f, 0.f};
  a01.v = a00.v; a10.v = a00.v; a11.v = a00.v;

  for (int k0 = 0; k0 < KPAD; k0 += 32) {
    BFrag fa0, fa1, fb0, fb1;
    fa0.u[0] = *(const v4u*)(Ap0 + k0);
    fa0.u[1] = *(const v4u*)(Ap0 + k0 + 16);
    fa1.u[0] = *(const v4u*)(Ap1 + k0);
    fa1.u[1] = *(const v4u*)(Ap1 + k0 + 16);
    fb0.u[0] = *(const v4u*)(Bp0 + k0);
    fb0.u[1] = *(const v4u*)(Bp0 + k0 + 16);
    fb1.u[0] = *(const v4u*)(Bp1 + k0);
    fb1.u[1] = *(const v4u*)(Bp1 + k0 + 16);
    __builtin_prefetch(Ap0 + k0 + 32, 0, 3);  // WGP-scope global_prefetch_b8
    __builtin_prefetch(Ap1 + k0 + 32, 0, 3);
    a00.v = __builtin_amdgcn_wmma_f32_16x16x32_bf16(false, fa0.v, false, fb0.v,
                                                    (short)0, a00.v, false, false);
    a01.v = __builtin_amdgcn_wmma_f32_16x16x32_bf16(false, fa0.v, false, fb1.v,
                                                    (short)0, a01.v, false, false);
    a10.v = __builtin_amdgcn_wmma_f32_16x16x32_bf16(false, fa1.v, false, fb0.v,
                                                    (short)0, a10.v, false, false);
    a11.v = __builtin_amdgcn_wmma_f32_16x16x32_bf16(false, fa1.v, false, fb1.v,
                                                    (short)0, a11.v, false, false);
  }

  // C/D layout: VGPR r, lanes 0-15 -> M=r, lanes 16-31 -> M=8+r; N=lane&15.
  const int rowBase = m0 + ((lane >> 4) << 3);
  const int nA = n0 + (lane & 15);
  const int nB = nA + 16;
  const float scA = (nA < Nout) ? sc[nA] : 0.f;
  const float shA = (nA < Nout) ? sh[nA] : 0.f;
  const float scB = (nB < Nout) ? sc[nB] : 0.f;
  const float shB = (nB < Nout) ? sh[nB] : 0.f;
#pragma unroll
  for (int r = 0; r < 8; ++r) {
    float v00 = fmaxf(a00.f[r] * scA + shA, 0.f);
    float v01 = fmaxf(a01.f[r] * scB + shB, 0.f);
    float v10 = fmaxf(a10.f[r] * scA + shA, 0.f);
    float v11 = fmaxf(a11.f[r] * scB + shB, 0.f);
    if (nA >= Nout) { v00 = 0.f; v10 = 0.f; }   // keep pad columns zero for next K
    if (nB >= Nout) { v01 = 0.f; v11 = 0.f; }
    size_t ro0 = (size_t)(rowBase + r) * NPAD;
    size_t ro1 = (size_t)(rowBase + r + 16) * NPAD;
    Out[ro0 + nA] = (__bf16)v00;
    Out[ro0 + nB] = (__bf16)v01;
    Out[ro1 + nA] = (__bf16)v10;
    Out[ro1 + nB] = (__bf16)v11;
  }
}

// ---------------------------------------------------------------------------
// Host orchestration
// ---------------------------------------------------------------------------
static inline int ru32h(int x) { return (x + 31) & ~31; }
static inline int gdiv(long long a, int b) { return (int)((a + b - 1) / b); }

extern "C" void kernel_launch(void* const* d_in, const int* in_sizes, int n_in,
                              void* d_out, int out_size, void* d_ws, size_t ws_size,
                              hipStream_t stream) {
  (void)in_sizes; (void)n_in; (void)out_size; (void)ws_size;
  const int B = 2, N = 16384;
  static const int NPl[5] = {16384, 2048, 1024, 512, 128};
  static const float SA_R[4][2] = {{0.0175f, 0.025f}, {0.025f, 0.05f},
                                   {0.05f, 0.1f},     {0.1f, 0.2f}};
  static const int SA_NS[4][2] = {{16, 32}, {16, 32}, {16, 32}, {16, 32}};
  static const int SA_CH[4][2][4] = {
      {{9, 16, 16, 32},     {9, 32, 32, 64}},
      {{99, 64, 64, 128},   {99, 64, 96, 128}},
      {{259, 128, 196, 256},{259, 128, 196, 256}},
      {{515, 256, 256, 512},{515, 256, 384, 512}}};
  static const int FP_CH[4][3] = {{262, 128, 128}, {608, 256, 256},
                                  {768, 512, 512}, {1536, 512, 512}};

  // ---- parse parameter pointers (setup_inputs flattening order) ----
  const float* pc = (const float*)d_in[0];
  int p = 1;
  const float *saW[4][2][3], *saS[4][2][3], *saH[4][2][3];
  for (int l = 0; l < 4; ++l)
    for (int s = 0; s < 2; ++s)
      for (int j = 0; j < 3; ++j) {
        saW[l][s][j] = (const float*)d_in[p++];
        saS[l][s][j] = (const float*)d_in[p++];
        saH[l][s][j] = (const float*)d_in[p++];
      }
  const float *fpWt[4][2], *fpS[4][2], *fpH[4][2];
  for (int l = 0; l < 4; ++l)
    for (int j = 0; j < 2; ++j) {
      fpWt[l][j] = (const float*)d_in[p++];
      fpS[l][j]  = (const float*)d_in[p++];
      fpH[l][j]  = (const float*)d_in[p++];
    }

  // ---- workspace bump allocator ----
  char* wsb = (char*)d_ws;
  size_t off = 0;
  auto alloc = [&](size_t bytes) -> void* {
    void* r = wsb + off;
    off = (off + bytes + 255) & ~(size_t)255;
    return r;
  };

  float* xyz[5];
  __bf16* feats[5];
  static const int featCap[5] = {128, 256, 512, 512, 1024};
  for (int l = 0; l < 5; ++l) {
    xyz[l] = (float*)alloc((size_t)B * NPl[l] * 3 * sizeof(float));
    feats[l] = (__bf16*)alloc((size_t)B * NPl[l] * featCap[l] * sizeof(__bf16));
  }
  float* fps_dist = (float*)alloc((size_t)B * N * sizeof(float));
  int* fps_idx    = (int*)alloc((size_t)B * 2048 * sizeof(int));
  int* ball       = (int*)alloc((size_t)B * 2048 * 32 * sizeof(int));
  int* idx3       = (int*)alloc((size_t)B * N * 3 * sizeof(int));
  float* w3       = (float*)alloc((size_t)B * N * 3 * sizeof(float));
  const size_t ACT_BYTES = (size_t)20 * 1024 * 1024;
  __bf16* actA = (__bf16*)alloc(ACT_BYTES);
  __bf16* actB = (__bf16*)alloc(ACT_BYTES);

  // ---- pad/convert weights to bf16 (per call; deterministic) ----
  __bf16* saWp[4][2][3];
  __bf16* fpWp[4][2];
  for (int l = 0; l < 4; ++l)
    for (int s = 0; s < 2; ++s)
      for (int j = 0; j < 3; ++j) {
        int cin = SA_CH[l][s][j], cout = SA_CH[l][s][j + 1];
        int kp = ru32h(cin), np = ru32h(cout);
        saWp[l][s][j] = (__bf16*)alloc((size_t)kp * np * sizeof(__bf16));
        int tot = kp * np;
        pad_weight_kernel<<<gdiv(tot, 256), 256, 0, stream>>>(saW[l][s][j], cin, cout, kp, np,
                                                              saWp[l][s][j]);
      }
  for (int l = 0; l < 4; ++l)
    for (int j = 0; j < 2; ++j) {
      int cin = FP_CH[l][j], cout = FP_CH[l][j + 1];
      int kp = ru32h(cin), np = ru32h(cout);
      fpWp[l][j] = (__bf16*)alloc((size_t)kp * np * sizeof(__bf16));
      int tot = kp * np;
      pad_weight_kernel<<<gdiv(tot, 256), 256, 0, stream>>>(fpWt[l][j], cin, cout, kp, np,
                                                            fpWp[l][j]);
    }

  auto gemm = [&](const __bf16* Ain, const __bf16* Wp, const float* scp, const float* shp,
                  __bf16* Aout, int M, int cout, int KPAD, int NPAD) {
    int tiles = (M / 32) * (NPAD / 32);
    wmma_mlp_kernel<<<gdiv(tiles, 4), 128, 0, stream>>>(Ain, Wp, scp, shp, Aout,
                                                        M, cout, NPAD, KPAD);
  };

  // ---- split pointcloud ----
  int featC[5];
  featC[0] = 6;
  {
    int tot = B * N;
    split_pc_kernel<<<gdiv(tot, 256), 256, 0, stream>>>(pc, tot, xyz[0], feats[0]);
  }

  // ---- SA levels ----
  for (int l = 0; l < 4; ++l) {
    int Np = NPl[l], S = NPl[l + 1];
    fps_kernel<<<B, FPS_T, 0, stream>>>(xyz[l], Np, S, fps_idx, fps_dist);
    {
      int tot = B * S;
      gather_xyz_kernel<<<gdiv(tot, 256), 256, 0, stream>>>(xyz[l], fps_idx, Np, S, tot,
                                                            xyz[l + 1]);
    }
    int Ctot = SA_CH[l][0][3] + SA_CH[l][1][3];
    int coff = 0;
    for (int s = 0; s < 2; ++s) {
      int ns = SA_NS[l][s];
      float r2 = SA_R[l][s] * SA_R[l][s];
      {
        int tot = B * S;
        ball_query_kernel<<<gdiv(tot, 256), 256, 0, stream>>>(xyz[l], xyz[l + 1], Np, S, tot,
                                                              r2, ns, ball);
      }
      int M = B * S * ns;
      int KP0 = ru32h(SA_CH[l][s][0]);
      {
        long long tot = (long long)M * KP0;
        group_gather_kernel<<<gdiv(tot, 256), 256, 0, stream>>>(
            xyz[l], xyz[l + 1], feats[l], featC[l], Np, S, ns, KP0, ball, actA, tot);
      }
      __bf16* cur = actA;
      __bf16* nxt = actB;
      int kp = KP0;
      for (int j = 0; j < 3; ++j) {
        int cout = SA_CH[l][s][j + 1];
        int np = ru32h(cout);
        gemm(cur, saWp[l][s][j], saS[l][s][j], saH[l][s][j], nxt, M, cout, kp, np);
        __bf16* t = cur; cur = nxt; nxt = t;
        kp = np;
      }
      int cout = SA_CH[l][s][3];   // multiple of 32 -> kp == cout, contiguous
      {
        int tot = B * S * cout;
        maxpool_kernel<<<gdiv(tot, 256), 256, 0, stream>>>(cur, S, ns, kp, cout,
                                                           feats[l + 1], Ctot, coff, tot);
      }
      coff += cout;
    }
    featC[l + 1] = Ctot;
  }

  // ---- FP levels (order: (3<-4), (2<-3), (1<-2), (0<-1)) ----
  static const int FPU[4] = {3, 2, 1, 0};
  for (int t = 0; t < 4; ++t) {
    int u = FPU[t], k = u + 1;
    int Nu = NPl[u], Nk = NPl[k];
    {
      int tot = B * Nu;
      three_nn_kernel<<<gdiv(tot, 256), 256, 0, stream>>>(xyz[u], xyz[k], Nu, Nk, tot,
                                                          idx3, w3);
    }
    int Ck = featC[k], Cu = featC[u];
    int KP0 = ru32h(Ck + Cu);
    int M = B * Nu;
    {
      long long tot = (long long)M * KP0;
      fp_input_kernel<<<gdiv(tot, 256), 256, 0, stream>>>(idx3, w3, feats[k], Ck, feats[u],
                                                          Cu, Nu, Nk, KP0, actA, tot);
    }
    int c1 = FP_CH[u][1], c2 = FP_CH[u][2];
    gemm(actA, fpWp[u][0], fpS[u][0], fpH[u][0], actB, M, c1, KP0, ru32h(c1));
    gemm(actB, fpWp[u][1], fpS[u][1], fpH[u][1], feats[u], M, c2, ru32h(c1), ru32h(c2));
    featC[u] = c2;
  }

  // ---- final transpose (B,N,128) bf16 -> (B,128,N) f32 ----
  {
    int C = featC[0];
    int tot = B * C * N;
    transpose_out_kernel<<<gdiv(tot, 256), 256, 0, stream>>>(feats[0], N, C,
                                                             (float*)d_out, tot);
  }
}
